// Model_83519934038703
// MI455X (gfx1250) — matrix operand order repaired and checked by reading, not compiled, not run on hardware
//
#include <hip/hip_runtime.h>
#include <hip/hip_bf16.h>

// ---------------- problem constants (from reference) ----------------
#define N_NODES  50000
#define E_EDGES  800000
#define IN_CH    128
#define HID_CH   256
#define OUT_CH   40
#define MAX_ITER 30
#define PHANTOM  5
#define TOL      3e-6f

typedef __attribute__((ext_vector_type(16))) __bf16 v16bf;
typedef __attribute__((ext_vector_type(8)))  float  v8f;

union ABFrag { v16bf v; uint4 q[2]; unsigned int u[8]; };

__device__ __forceinline__ void atomicAddF32(float* p, float v) {
    unsafeAtomicAdd(p, v);   // native global_atomic_add_f32 on gfx1250
}

// ---------------- f32 -> bf16 conversion (optionally fused relu) ----------------
__global__ void k_cvt_bf16(const float* __restrict__ in, __bf16* __restrict__ out,
                           int n, int do_relu) {
    int i = blockIdx.x * blockDim.x + threadIdx.x;
    int stride = gridDim.x * blockDim.x;
    for (; i < n; i += stride) {
        float v = in[i];
        if (do_relu) v = fmaxf(v, 0.0f);
        out[i] = (__bf16)v;
    }
}

__global__ void k_fill_f32(float* __restrict__ p, float v, int n) {
    int i = blockIdx.x * blockDim.x + threadIdx.x;
    int stride = gridDim.x * blockDim.x;
    for (; i < n; i += stride) p[i] = v;
}

__global__ void k_init_flag(int* flag) { *flag = 0; }

// ------- WMMA bf16 GEMM:  C[M,Nd] = scale * (A[M,K] @ W[Nd,K]^T) + bias ----------
// One 16x(16*NT) tile strip per wave: the A fragment is loaded once per K-step
// and reused across NT column tiles. All NT B fragments of a K-step are issued
// as one load batch BEFORE the wmma sequence, so each v_wmma waits only for its
// own operands (staggered s_wait_loadcnt) while later loads stay in flight.
// K is compile-time so the K-loop fully unrolls; all fragment loads are b128.
template <int K, int NT, bool OUT_BF16, bool HAS_BIAS, bool HAS_SCALE>
__global__ void k_wmma_gemm(const __bf16* __restrict__ A, int lda,
                            const __bf16* __restrict__ W, int ldw,
                            float* __restrict__ Cf, __bf16* __restrict__ Cbf, int ldc,
                            int M, int Nd,
                            const float* __restrict__ bias,
                            const float* __restrict__ scale_p) {
    const int lane = threadIdx.x & 31;
    const int half = lane >> 4;      // K-half select (wave32 layout)
    const int l16  = lane & 15;      // M row (A) / N col (B) within tile
    int wid = (blockIdx.x * blockDim.x + threadIdx.x) >> 5;
    int nw  = (gridDim.x * blockDim.x) >> 5;
    const int ntN  = (Nd + 15) >> 4;
    const int ngN  = ntN / NT;                  // ntN divisible by NT (by choice)
    const int ngroups = (M >> 4) * ngN;         // M is a multiple of 16 here
    const float scale = HAS_SCALE ? *scale_p : 1.0f;

    for (int t = wid; t < ngroups; t += nw) {
        const int mb = (t / ngN) << 4;
        const int nb = (t % ngN) * (16 * NT);
        const int arow = mb + l16;
        const __bf16* ap = A + arow * lda;      // 32-bit offset math

        int ncol[NT];
        const __bf16* wp[NT];
#pragma unroll
        for (int j = 0; j < NT; ++j) {
            ncol[j] = nb + 16 * j + l16;
            int ncl = ncol[j] < Nd ? ncol[j] : (Nd - 1);   // clamp: no predication
            wp[j] = W + ncl * ldw;
        }

        v8f acc[NT];
#pragma unroll
        for (int j = 0; j < NT; ++j) acc[j] = v8f{};

#pragma unroll
        for (int kb = 0; kb < K; kb += 32) {
            // A 16x32 bf16 fragment: lane-half selects K {0..7,16..23} vs
            // {8..15,24..31}; each run of 8 K-values is one b128 load.
            ABFrag a;
            a.q[0] = *(const uint4*)(ap + kb + half * 8);
            a.q[1] = *(const uint4*)(ap + kb + 16 + half * 8);
            // Batch all NT B fragments (same K split as A = 2 b128):
            ABFrag b[NT];
#pragma unroll
            for (int j = 0; j < NT; ++j) {
                b[j].q[0] = *(const uint4*)(wp[j] + kb + half * 8);
                b[j].q[1] = *(const uint4*)(wp[j] + kb + 16 + half * 8);
            }
            // Back-to-back XDL ops; waits stagger across the load batch.
#pragma unroll
            for (int j = 0; j < NT; ++j) {
                acc[j] = __builtin_amdgcn_wmma_f32_16x16x32_bf16(
                             false, a.v, false, b[j].v, (short)0, acc[j], false, false);
            }
        }

        // C/D layout: VGPR i -> row mb + half*8 + i, col = ncol[j]
#pragma unroll
        for (int j = 0; j < NT; ++j) {
            const float bv = HAS_BIAS ? bias[ncol[j] < Nd ? ncol[j] : (Nd - 1)] : 0.0f;
            if (ncol[j] < Nd) {                 // single exec-mask for 8 stores
#pragma unroll
                for (int i = 0; i < 8; ++i) {
                    int r = mb + half * 8 + i;
                    float val = acc[j][i] * scale + bv;
                    int o = r * ldc + ncol[j];
                    if (OUT_BF16) Cbf[o] = (__bf16)val;
                    else          Cf[o]  = val;
                }
            }
        }
    }
}

// ---------------- seed accumulator: z_out = u / gamma ; also zero norm accums ----
__global__ void k_seed_acc(float4* __restrict__ z_out, const float4* __restrict__ u,
                           const float* __restrict__ gamma_p, float* __restrict__ norms,
                           const int* __restrict__ flag, int respect, int n4) {
    if (blockIdx.x == 0 && threadIdx.x == 0) { norms[0] = 0.0f; norms[1] = 0.0f; }
    if (respect && *flag) return;
    const float inv_g = 1.0f / *gamma_p;
    int i = blockIdx.x * blockDim.x + threadIdx.x;
    int stride = gridDim.x * blockDim.x;
    for (; i < n4; i += stride) {
        float4 v = u[i];
        v.x *= inv_g; v.y *= inv_g; v.z *= inv_g; v.w *= inv_g;
        z_out[i] = v;
    }
}

// ---------------- SpMM scatter: z_acc[dst] += w * z_in[src]  (one wave per edge) ----
__global__ void k_spmm_scatter(const float* __restrict__ z_in, float* __restrict__ z_acc,
                               const long long* __restrict__ src,
                               const long long* __restrict__ dst,
                               const float* __restrict__ w,
                               const int* __restrict__ flag, int respect, int nedges) {
    if (respect && *flag) return;
    const int lane = threadIdx.x & 31;
    int wid = (blockIdx.x * blockDim.x + threadIdx.x) >> 5;
    int nw  = (gridDim.x * blockDim.x) >> 5;
    for (int e = wid; e < nedges; e += nw) {
        if (e + nw < nedges) {       // global_prefetch_b8 for next edge record
            __builtin_prefetch(&src[e + nw], 0, 1);
            __builtin_prefetch(&dst[e + nw], 0, 1);
            __builtin_prefetch(&w[e + nw], 0, 1);
        }
        int soff = (int)src[e] * HID_CH;
        int doff = (int)dst[e] * HID_CH;
        float we = w[e];
        const float4* zr4 = (const float4*)(z_in + soff);   // row is 16B aligned
        float* za = z_acc + doff;
        // 256 floats/row: each lane owns two float4 slots -> 2 b128 loads,
        // 8 native f32 atomics (the f32 scatter floor on CDNA5).
        float4 v0 = zr4[lane];
        float4 v1 = zr4[lane + 32];
        float* za0 = za + lane * 4;
        float* za1 = za + (lane + 32) * 4;
        atomicAddF32(za0 + 0, we * v0.x);
        atomicAddF32(za0 + 1, we * v0.y);
        atomicAddF32(za0 + 2, we * v0.z);
        atomicAddF32(za0 + 3, we * v0.w);
        atomicAddF32(za1 + 0, we * v1.x);
        atomicAddF32(za1 + 1, we * v1.y);
        atomicAddF32(za1 + 2, we * v1.z);
        atomicAddF32(za1 + 3, we * v1.w);
    }
}

// ---------------- update: z = relu(gamma * acc) in place; residual norms ----------
__global__ void k_relu_update(float4* __restrict__ z_io, const float4* __restrict__ z_prev,
                              const float* __restrict__ gamma_p, float* __restrict__ norms,
                              const int* __restrict__ flag, int respect, int track, int n4) {
    const bool frozen = respect && (*flag != 0);
    const float g = *gamma_p;
    float s_d = 0.0f, s_n = 0.0f;
    int i = blockIdx.x * blockDim.x + threadIdx.x;
    int stride = gridDim.x * blockDim.x;
    for (; i < n4; i += stride) {
        float4 zp = z_prev[i];
        float4 v;
        if (frozen) {
            v = zp;
        } else {
            float4 acc = z_io[i];
            v.x = fmaxf(g * acc.x, 0.0f);
            v.y = fmaxf(g * acc.y, 0.0f);
            v.z = fmaxf(g * acc.z, 0.0f);
            v.w = fmaxf(g * acc.w, 0.0f);
        }
        z_io[i] = v;
        float dx = v.x - zp.x, dy = v.y - zp.y, dz = v.z - zp.z, dw = v.w - zp.w;
        s_d += dx * dx + dy * dy + dz * dz + dw * dw;
        s_n += v.x * v.x + v.y * v.y + v.z * v.z + v.w * v.w;
    }
    if (track && !frozen) {
        for (int o = 16; o > 0; o >>= 1) {     // wave32 reduction
            s_d += __shfl_down(s_d, o, 32);
            s_n += __shfl_down(s_n, o, 32);
        }
        if ((threadIdx.x & 31) == 0) {
            atomicAddF32(&norms[0], s_d);
            atomicAddF32(&norms[1], s_n);
        }
    }
}

__global__ void k_check(const float* __restrict__ norms, int* __restrict__ flag) {
    if (*flag) return;
    float diff = sqrtf(norms[0]) / (sqrtf(norms[1]) + 1e-12f);
    if (!(diff > TOL)) *flag = 1;
}

// =============================== launcher =====================================
extern "C" void kernel_launch(void* const* d_in, const int* in_sizes, int n_in,
                              void* d_out, int out_size, void* d_ws, size_t ws_size,
                              hipStream_t stream) {
    (void)in_sizes; (void)n_in; (void)out_size; (void)ws_size;
    const float*     x      = (const float*)d_in[0];
    const long long* eidx   = (const long long*)d_in[1];   // int64 [2,E]
    const float*     ew     = (const float*)d_in[2];
    const float*     W_enc  = (const float*)d_in[3];
    const float*     b_enc  = (const float*)d_in[4];
    const float*     W_bias = (const float*)d_in[5];
    const float*     W_dec  = (const float*)d_in[6];
    const float*     b_dec  = (const float*)d_in[7];
    const float*     beta   = (const float*)d_in[8];
    const float*     gamma  = (const float*)d_in[9];
    const long long* src = eidx;
    const long long* dst = eidx + E_EDGES;
    float* out = (float*)d_out;

    // ---- workspace carve (all 256B aligned) ----
    char* p = (char*)d_ws;
    auto carve = [&](size_t bytes) {
        char* r = p; p += (bytes + 255) & ~(size_t)255; return r;
    };
    const int NH  = N_NODES * HID_CH;
    const int NH4 = NH / 4;
    float*  u   = (float*)carve(sizeof(float) * NH);
    float*  zA  = (float*)carve(sizeof(float) * NH);
    float*  zB  = (float*)carve(sizeof(float) * NH);
    __bf16* x_bf  = (__bf16*)carve(sizeof(__bf16) * (size_t)N_NODES * (IN_CH + HID_CH));
    __bf16* h_bf  = x_bf + (size_t)N_NODES * IN_CH;
    __bf16* zr_bf = x_bf;                       // reused after GEMM1/2 are done
    __bf16* wenc_bf  = (__bf16*)carve(sizeof(__bf16) * HID_CH * IN_CH);
    __bf16* wbias_bf = (__bf16*)carve(sizeof(__bf16) * HID_CH * HID_CH);
    __bf16* wdec_bf  = (__bf16*)carve(sizeof(__bf16) * OUT_CH * HID_CH);
    float*  norms = (float*)carve(256);
    int*    flag  = (int*)carve(256);

    const int T = 256;
    k_init_flag<<<1, 1, 0, stream>>>(flag);

    // bf16 copies of activations + weights
    k_cvt_bf16<<<(N_NODES*IN_CH + T-1)/T, T, 0, stream>>>(x, x_bf, N_NODES*IN_CH, 0);
    k_cvt_bf16<<<(HID_CH*IN_CH  + T-1)/T, T, 0, stream>>>(W_enc,  wenc_bf,  HID_CH*IN_CH, 0);
    k_cvt_bf16<<<(HID_CH*HID_CH + T-1)/T, T, 0, stream>>>(W_bias, wbias_bf, HID_CH*HID_CH, 0);
    k_cvt_bf16<<<(OUT_CH*HID_CH + T-1)/T, T, 0, stream>>>(W_dec,  wdec_bf,  OUT_CH*HID_CH, 0);

    // GEMM1: h = x @ W_enc^T + b_enc  (bf16 out, bias, no scale) NT=4
    {
        int ngroups = (N_NODES / 16) * ((HID_CH / 16) / 4);
        int blocks  = (ngroups + 7) / 8;                  // 8 waves / block
        k_wmma_gemm<IN_CH, 4, true, true, false>
            <<<blocks, T, 0, stream>>>(x_bf, IN_CH, wenc_bf, IN_CH,
                                       nullptr, h_bf, HID_CH,
                                       N_NODES, HID_CH, b_enc, nullptr);
    }
    // GEMM2: u = beta * (h @ W_bias^T)  (f32 out, no bias, scale) NT=4
    {
        int ngroups = (N_NODES / 16) * ((HID_CH / 16) / 4);
        int blocks  = (ngroups + 7) / 8;
        k_wmma_gemm<HID_CH, 4, false, false, true>
            <<<blocks, T, 0, stream>>>(h_bf, HID_CH, wbias_bf, HID_CH,
                                       u, nullptr, HID_CH,
                                       N_NODES, HID_CH, nullptr, beta);
    }

    // z0 = 0
    k_fill_f32<<<(NH + T-1)/T, T, 0, stream>>>(zA, 0.0f, NH);

    float* cur = zA;
    float* nxt = zB;
    // ---- fixed-point solve: 30 flag-gated iterations (graph-capture safe) ----
    for (int it = 0; it < MAX_ITER; ++it) {
        k_seed_acc    <<<4096, T, 0, stream>>>((float4*)nxt, (const float4*)u, gamma,
                                               norms, flag, 1, NH4);
        k_spmm_scatter<<<2048, T, 0, stream>>>(cur, nxt, src, dst, ew, flag, 1, E_EDGES);
        k_relu_update <<<4096, T, 0, stream>>>((float4*)nxt, (const float4*)cur, gamma,
                                               norms, flag, 1, 1, NH4);
        k_check       <<<1, 1, 0, stream>>>(norms, flag);
        float* t2 = cur; cur = nxt; nxt = t2;
    }
    // ---- phantom iterations: always run ----
    for (int it = 0; it < PHANTOM; ++it) {
        k_seed_acc    <<<4096, T, 0, stream>>>((float4*)nxt, (const float4*)u, gamma,
                                               norms, flag, 0, NH4);
        k_spmm_scatter<<<2048, T, 0, stream>>>(cur, nxt, src, dst, ew, flag, 0, E_EDGES);
        k_relu_update <<<4096, T, 0, stream>>>((float4*)nxt, (const float4*)cur, gamma,
                                               norms, flag, 0, 0, NH4);
        float* t2 = cur; cur = nxt; nxt = t2;
    }

    // decoder: out = relu(z) @ W_dec^T + b_dec  (f32 out, bias) NT=3 covers 48 cols
    k_cvt_bf16<<<(NH + T-1)/T, T, 0, stream>>>(cur, zr_bf, NH, 1);
    {
        int ngroups = (N_NODES / 16) * 1;                 // ntN=3, NT=3 -> 1 group/row
        int blocks  = (ngroups + 7) / 8;
        k_wmma_gemm<HID_CH, 3, false, true, false>
            <<<blocks, T, 0, stream>>>(zr_bf, HID_CH, wdec_bf, HID_CH,
                                       out, nullptr, OUT_CH,
                                       N_NODES, OUT_CH, b_dec, nullptr);
    }
}
